// Encoder_75230647156738
// MI455X (gfx1250) — matrix operand order repaired and checked by reading, not compiled, not run on hardware
//
#include <hip/hip_runtime.h>
#include <hip/hip_bf16.h>
#include <stdint.h>
#include <stddef.h>

// Problem sizes (fixed by the reference)
#define SEQ_LEN 4096
#define INSZ    512
#define H       2048
#define FH      8192   // 4*H

typedef __attribute__((ext_vector_type(16))) __bf16 bf16x16;
typedef __attribute__((ext_vector_type(8)))  float  f32x8;

// ---------------- fp32 -> bf16 (round-to-nearest-even) ----------------
__global__ void cvt_f32_to_bf16(const float* __restrict__ in,
                                uint16_t* __restrict__ out, int n) {
  int i = blockIdx.x * blockDim.x + threadIdx.x;
  if (i < n) {
    uint32_t u = __float_as_uint(in[i]);
    u += 0x7FFFu + ((u >> 16) & 1u);
    out[i] = (uint16_t)(u >> 16);
  }
}

// ---------------- Phase 1: x_proj = src @ W_ih^T + b_ih  (bf16 WMMA) ----------------
// One wave computes one 16x16 output tile, looping K=512 in steps of 32.
// Fragment layouts per CDNA5 ISA 7.12.2 (wave32):
//   A (16x32 bf16): lanes 0-15 -> M=lane, K={0..7,16..23}; lanes 16-31 -> K={8..15,24..31}
//   B (32x16 bf16): lane holds col n=lane&15; lanes 0-15 -> K={0..7,16..23}, lanes 16-31 -> K={8..15,24..31}
//   C/D (16x16 f32): VGPR v -> row v (+8 for upper lane half), col = lane&15
union Frag32B { uint4 q[2]; bf16x16 v; };

__global__ __launch_bounds__(256)
void xproj_wmma(const uint16_t* __restrict__ Abf,   // src_bf16 [SEQ, INSZ]
                const uint16_t* __restrict__ Bbf,   // W_ih_bf16 [FH, INSZ]
                const float*    __restrict__ bias,  // b_ih [FH]
                float*          __restrict__ xproj) // [SEQ, FH]
{
  const int lane  = threadIdx.x & 31;
  const int wave  = threadIdx.x >> 5;
  const int ntile = blockIdx.x;                 // 0 .. FH/16-1   (shared B across waves)
  const int mtile = blockIdx.y * 8 + wave;      // 0 .. SEQ/16-1
  const int mn    = lane & 15;
  const int sel   = lane >> 4;                  // which K half this lane holds

  const uint16_t* arow = Abf + (size_t)(mtile * 16 + mn) * INSZ;
  const uint16_t* brow = Bbf + (size_t)(ntile * 16 + mn) * INSZ;

  f32x8 c = {};
#pragma unroll
  for (int k0 = 0; k0 < INSZ; k0 += 32) {
    Frag32B a, b;
    a.q[0] = *(const uint4*)(arow + k0 +      sel * 8);   // K 0..7  / 8..15
    a.q[1] = *(const uint4*)(arow + k0 + 16 + sel * 8);   // K 16..23 / 24..31
    b.q[0] = *(const uint4*)(brow + k0 +      sel * 8);   // K 0..7  / 8..15
    b.q[1] = *(const uint4*)(brow + k0 + 16 + sel * 8);   // K 16..23 / 24..31
    c = __builtin_amdgcn_wmma_f32_16x16x32_bf16(
            /*neg_a=*/false, a.v, /*neg_b=*/false, b.v,
            /*c_mod=*/(short)0, c, /*reuse_a=*/false, /*reuse_b=*/false);
  }

  const float bv = bias[ntile * 16 + mn];
  float* orow = xproj + (size_t)(mtile * 16 + sel * 8) * FH + ntile * 16 + mn;
#pragma unroll
  for (int v = 0; v < 8; ++v)
    orow[(size_t)v * FH] = c[v] + bv;
}

// ---------------- Phase 2: serial LSTM scan ----------------
// 2048 waves (64 blocks x 1024 threads), one hidden unit j per wave.
// W_hh kept bf16 (32 MB -> L2-resident). h ping-pong in global, broadcast to LDS.
// c[j] lives in a lane-0 register for the whole scan.
#define NBLK 64
#define TPB  1024

__device__ __forceinline__ float wave_reduce(float a) {
#pragma unroll
  for (int off = 16; off; off >>= 1) a += __shfl_xor(a, off, 32);
  return a;
}

__global__ __launch_bounds__(TPB)
void lstm_scan(const float*    __restrict__ xproj, // [SEQ, FH]
               const uint16_t* __restrict__ Whh,   // [FH, H] bf16
               const float*    __restrict__ bhh,   // [FH]
               float*          __restrict__ hbuf,  // [2, H] ping-pong
               unsigned*       __restrict__ bar,   // [0]=count [1]=generation
               float*          __restrict__ out)   // [2*H] = h, c
{
  __shared__ float hs[H];
  const int lane = threadIdx.x & 31;
  const int wave = threadIdx.x >> 5;
  const int j    = blockIdx.x * (TPB / 32) + wave;   // hidden unit 0..H-1

  const uint16_t* w0 = Whh + (size_t)(0 * H + j) * H;
  const uint16_t* w1 = Whh + (size_t)(1 * H + j) * H;
  const uint16_t* w2 = Whh + (size_t)(2 * H + j) * H;
  const uint16_t* w3 = Whh + (size_t)(3 * H + j) * H;

  float c = 0.0f;
  float bi = 0.f, bff = 0.f, bg = 0.f, bo = 0.f;
  if (lane == 0) {
    bi  = bhh[j];
    bff = bhh[H + j];
    bg  = bhh[2 * H + j];
    bo  = bhh[3 * H + j];
  }

  for (int t = 0; t < SEQ_LEN; ++t) {
    const float* hr = hbuf + (size_t)(t & 1) * H;        // read h_{t}
    float*       hw = hbuf + (size_t)((t & 1) ^ 1) * H;  // write h_{t+1}

    // broadcast h into LDS (block-wide)
    for (int i = threadIdx.x; i < H; i += TPB) hs[i] = hr[i];
    __syncthreads();

    float a0 = 0.f, a1 = 0.f, a2 = 0.f, a3 = 0.f;
#pragma unroll 4
    for (int kk = 0; kk < H / 64; ++kk) {
      const int k = kk * 64 + 2 * lane;     // lane-contiguous bf16 pairs
      const float h0 = hs[k], h1 = hs[k + 1];
      uint32_t u;
      u = *(const uint32_t*)(w0 + k);
      a0 = fmaf(__uint_as_float(u << 16), h0, a0);
      a0 = fmaf(__uint_as_float(u & 0xFFFF0000u), h1, a0);
      u = *(const uint32_t*)(w1 + k);
      a1 = fmaf(__uint_as_float(u << 16), h0, a1);
      a1 = fmaf(__uint_as_float(u & 0xFFFF0000u), h1, a1);
      u = *(const uint32_t*)(w2 + k);
      a2 = fmaf(__uint_as_float(u << 16), h0, a2);
      a2 = fmaf(__uint_as_float(u & 0xFFFF0000u), h1, a2);
      u = *(const uint32_t*)(w3 + k);
      a3 = fmaf(__uint_as_float(u << 16), h0, a3);
      a3 = fmaf(__uint_as_float(u & 0xFFFF0000u), h1, a3);
    }
    a0 = wave_reduce(a0);
    a1 = wave_reduce(a1);
    a2 = wave_reduce(a2);
    a3 = wave_reduce(a3);

    if (lane == 0) {
      const float* xp = xproj + (size_t)t * FH;
      float gi = a0 + xp[j]         + bi;
      float gf = a1 + xp[H + j]     + bff;
      float gg = a2 + xp[2 * H + j] + bg;
      float go = a3 + xp[3 * H + j] + bo;
      gi = 1.0f / (1.0f + __expf(-gi));
      gf = 1.0f / (1.0f + __expf(-gf));
      go = 1.0f / (1.0f + __expf(-go));
      gg = tanhf(gg);
      c = fmaf(gf, c, gi * gg);
      const float hn = go * tanhf(c);
      hw[j] = hn;
      if (t == SEQ_LEN - 1) { out[j] = hn; out[H + j] = c; }
    }
    __threadfence();

    // device-wide barrier (generation counter, agent scope)
    __syncthreads();
    if (threadIdx.x == 0) {
      const unsigned g = __hip_atomic_load(bar + 1, __ATOMIC_RELAXED, __HIP_MEMORY_SCOPE_AGENT);
      const unsigned a = __hip_atomic_fetch_add(bar, 1u, __ATOMIC_ACQ_REL, __HIP_MEMORY_SCOPE_AGENT);
      if (a == NBLK - 1) {
        __hip_atomic_store(bar, 0u, __ATOMIC_RELAXED, __HIP_MEMORY_SCOPE_AGENT);
        __hip_atomic_fetch_add(bar + 1, 1u, __ATOMIC_RELEASE, __HIP_MEMORY_SCOPE_AGENT);
      } else {
        while (__hip_atomic_load(bar + 1, __ATOMIC_ACQUIRE, __HIP_MEMORY_SCOPE_AGENT) == g)
          __builtin_amdgcn_s_sleep(1);
      }
    }
    __syncthreads();
  }
}

// ---------------- workspace layout ----------------
#define OFF_XPROJ  ((size_t)0)
#define SZ_XPROJ   ((size_t)SEQ_LEN * FH * 4)                 // 128 MB
#define OFF_SRCBF  (OFF_XPROJ + SZ_XPROJ)
#define SZ_SRCBF   ((size_t)SEQ_LEN * INSZ * 2)               // 4 MB
#define OFF_WIHBF  (OFF_SRCBF + SZ_SRCBF)
#define SZ_WIHBF   ((size_t)FH * INSZ * 2)                    // 8 MB
#define OFF_WHHBF  (OFF_WIHBF + SZ_WIHBF)
#define SZ_WHHBF   ((size_t)FH * H * 2)                       // 32 MB
#define OFF_HBUF   (OFF_WHHBF + SZ_WHHBF)
#define SZ_HBUF    ((size_t)2 * H * 4)                        // 16 KB ping-pong
#define OFF_BAR    (OFF_HBUF + SZ_HBUF)
#define SZ_BAR     ((size_t)256)

extern "C" void kernel_launch(void* const* d_in, const int* in_sizes, int n_in,
                              void* d_out, int out_size, void* d_ws, size_t ws_size,
                              hipStream_t stream) {
  const float* src = (const float*)d_in[0];   // [SEQ, INSZ]
  const float* Wih = (const float*)d_in[1];   // [FH, INSZ]
  const float* Whh = (const float*)d_in[2];   // [FH, H]
  const float* bih = (const float*)d_in[3];   // [FH]
  const float* bhh = (const float*)d_in[4];   // [FH]
  float* out = (float*)d_out;                 // [2*H] = (h, c)

  char* ws = (char*)d_ws;
  float*    xproj = (float*)   (ws + OFF_XPROJ);
  uint16_t* srcbf = (uint16_t*)(ws + OFF_SRCBF);
  uint16_t* wihbf = (uint16_t*)(ws + OFF_WIHBF);
  uint16_t* whhbf = (uint16_t*)(ws + OFF_WHHBF);
  float*    hbuf  = (float*)   (ws + OFF_HBUF);
  unsigned* bar   = (unsigned*)(ws + OFF_BAR);

  // fp32 -> bf16 conversions (one-time, HBM-bound)
  {
    int n = SEQ_LEN * INSZ;
    cvt_f32_to_bf16<<<(n + 255) / 256, 256, 0, stream>>>(src, srcbf, n);
  }
  {
    int n = FH * INSZ;
    cvt_f32_to_bf16<<<(n + 255) / 256, 256, 0, stream>>>(Wih, wihbf, n);
  }
  {
    int n = FH * H;
    cvt_f32_to_bf16<<<(n + 255) / 256, 256, 0, stream>>>(Whh, whhbf, n);
  }

  // zero h ping-pong + barrier state every launch (graph-capture safe)
  hipMemsetAsync(ws + OFF_HBUF, 0, SZ_HBUF + SZ_BAR, stream);

  // Phase 1: WMMA GEMM. grid = (N tiles, M tiles / 8 waves)
  dim3 g1(FH / 16, SEQ_LEN / (16 * 8));
  xproj_wmma<<<g1, 256, 0, stream>>>(srcbf, wihbf, bih, xproj);

  // Phase 2: persistent serial scan
  lstm_scan<<<NBLK, TPB, 0, stream>>>(xproj, whhbf, bhh, hbuf, bar, out);
}
